// MLPDecoder_2052994367645
// MI455X (gfx1250) — matrix operand order repaired and checked by reading, not compiled, not run on hardware
//
#include <hip/hip_runtime.h>

// ---------------------------------------------------------------------------
// NRI MLPDecoder fused for MI455X (gfx1250, wave32, WMMA bf16 16x16x32).
//
//  * edges for receiver n are contiguous [n*127,(n+1)*127), sender j=m+(m>=n)
//  * the 2-iteration edge-type loop uses shared weights -> msg*(rt0+rt1)
//  * compute-bound (51 GFLOP vs ~7MB traffic) -> fuse, bf16 WMMA, f32 accum
//
// Round-2 changes:
//  * k_edge stages W1 (64K) + W2 (32K) bf16 weights into LDS with
//    GLOBAL_LOAD_ASYNC_TO_LDS_B128 (ASYNCcnt path, overlapped with the
//    fp32->bf16 pre_msg build), padded row strides for bank-conflict-free
//    A-fragment reads. Uses gfx1250's 320KB LDS (~140KB/block).
//
// WMMA operand mapping (ISA 7.12.2):
//  A (16x32 bf16): lane r holds row M=r; elems 0..7 = K(kt*32+h*8..),
//                  elems 8..15 = K(kt*32+16+h*8..)   -> two 16B loads
//  B (32x16 bf16): lane (r,h) holds col N=r, elems 0..7 = K(kt*32+h*8..), 8..15 = K(kt*32+16+h*8..)
//  C/D (16x16 f32): lane (r,h), reg v -> D[M=h*8+v][N=r]
// WEIGHTS in A (M=out channel), ACTIVATIONS in B (N=row): layer outputs sit
// per-lane as "all channels of my row"; the next layer's B fragment is
// rebuilt from the lane's own packed registers, no shuffle.
// ---------------------------------------------------------------------------

#define Bsz 32
#define Nn  128
#define NIN 64
#define Ee  (Nn * (Nn - 1))   // 16256

typedef __bf16        v16bf __attribute__((ext_vector_type(16)));
typedef float         v8f   __attribute__((ext_vector_type(8)));
typedef unsigned int  v8u   __attribute__((ext_vector_type(8)));
typedef unsigned int  v4u   __attribute__((ext_vector_type(4)));

__device__ __forceinline__ unsigned short f2bf(float f) {
  unsigned u = __builtin_bit_cast(unsigned, f);
  u += 0x7FFFu + ((u >> 16) & 1u);           // round-to-nearest-even
  return (unsigned short)(u >> 16);
}
__device__ __forceinline__ unsigned pack2(float a, float b) {
  return (unsigned)f2bf(a) | ((unsigned)f2bf(b) << 16);
}
__device__ __forceinline__ v16bf ld2x16(const unsigned short* p0,
                                        const unsigned short* p1) {
  v4u lo = *(const v4u*)p0;
  v4u hi = *(const v4u*)p1;
  v8u u = {lo.x, lo.y, lo.z, lo.w, hi.x, hi.y, hi.z, hi.w};
  return __builtin_bit_cast(v16bf, u);
}
__device__ __forceinline__ v8f wmma_bf16(v16bf a, v16bf b, v8f c) {
  return __builtin_amdgcn_wmma_f32_16x16x32_bf16(false, a, false, b,
                                                 (short)0, c, false, false);
}
__device__ __forceinline__ v8f zero8() {
  v8f z = {0.f, 0.f, 0.f, 0.f, 0.f, 0.f, 0.f, 0.f};
  return z;
}
__device__ __forceinline__ v16bf build_bfrag(const unsigned (&pk)[16][4],
                                             int kt, int h) {
  unsigned lo[4], hi[4];
#pragma unroll
  for (int d = 0; d < 4; ++d) {
    lo[d] = pk[2 * kt][d];
    hi[d] = pk[2 * kt + 1][d];
  }
  v8u bu = {lo[0], lo[1], lo[2], lo[3], hi[0], hi[1], hi[2], hi[3]};
  return __builtin_bit_cast(v16bf, bu);
}

// --- gfx1250 async global->LDS copy (ASYNCcnt path, ISA ch.10 / 15.18) -----
__device__ __forceinline__ unsigned lds_off32(const void* p) {
  // generic LDS pointer: addr[31:0] is the LDS byte offset (ISA 10.2)
  return (unsigned)(uintptr_t)p;
}
__device__ __forceinline__ void async_g2l_b128(unsigned lds_byte_off,
                                               const void* gptr) {
  asm volatile("global_load_async_to_lds_b128 %0, %1, off"
               :: "v"(lds_byte_off), "v"(gptr)
               : "memory");
}
__device__ __forceinline__ void wait_async0() {
#if __has_builtin(__builtin_amdgcn_s_wait_asynccnt)
  __builtin_amdgcn_s_wait_asynccnt(0);
#else
  asm volatile("s_wait_asynccnt 0x0" ::: "memory");
#endif
}

// ---------------------------------------------------------------------------
// Kernel 0: fp32 -> bf16 weight conversion into workspace (L2-resident).
// Segments (elements): W1 32768 | W2 16384 | W3 16384 | W4 65536 | W5 16384
// ---------------------------------------------------------------------------
__global__ void k_convert(const float* __restrict__ w1, const float* __restrict__ w2,
                          const float* __restrict__ w3, const float* __restrict__ w4,
                          const float* __restrict__ w5, unsigned short* __restrict__ out) {
  int i = blockIdx.x * 256 + threadIdx.x;
  if (i >= 147456) return;
  float v;
  if      (i <  32768) v = w1[i];
  else if (i <  49152) v = w2[i - 32768];
  else if (i <  65536) v = w3[i - 49152];
  else if (i < 131072) v = w4[i - 65536];
  else                 v = w5[i - 131072];
  out[i] = f2bf(v);
}

// ---------------------------------------------------------------------------
// Kernel 1: edge MLP + scale + edge2node aggregation.
// grid = (N=128, B=32), 256 threads (8 waves, wave w owns 16 edge rows).
// ---------------------------------------------------------------------------
__global__ __launch_bounds__(256, 1) void k_edge(
    const float* __restrict__ inputs,            // [B,128,64]
    const float* __restrict__ rel_types,         // [B,E,2]
    const unsigned short* __restrict__ W1g,      // [256,128] bf16
    const unsigned short* __restrict__ W2g,      // [64,256]  bf16
    const float* __restrict__ b1,                // [256]
    const float* __restrict__ b2,                // [64]
    float* __restrict__ aggW) {                  // [B*128,64]
  constexpr int PS  = 136;                       // pre_msg row stride (128+8)
  constexpr int W1S = 136;                       // staged W1 row stride (128+8)
  constexpr int W2S = 264;                       // staged W2 row stride (256+8)
  __shared__ alignas(16) unsigned short sPre[128 * PS];   // 34.8 KB
  __shared__ alignas(16) unsigned short sW1[256 * W1S];   // 69.6 KB
  __shared__ alignas(16) unsigned short sW2[64 * W2S];    // 33.8 KB
  __shared__ float sB1[256];
  __shared__ float sB2[64];
  __shared__ float sAgg[64];

  const int n   = blockIdx.x;
  const int b   = blockIdx.y;
  const int tid = threadIdx.x;
  const int lane = tid & 31;
  const int w   = tid >> 5;
  const int r   = lane & 15;
  const int h   = lane >> 4;

  // --- kick off async weight staging: 16B chunks, padded row placement ---
  {
    const unsigned w1base = lds_off32(sW1);
#pragma unroll
    for (int i = 0; i < 16; ++i) {               // 4096 chunks of W1
      int c = i * 256 + tid;
      int row = c >> 4, col = c & 15;            // 16 chunks per 128-col row
      async_g2l_b128(w1base + (unsigned)(row * W1S + col * 8) * 2, W1g + c * 8);
    }
    const unsigned w2base = lds_off32(sW2);
#pragma unroll
    for (int i = 0; i < 8; ++i) {                // 2048 chunks of W2
      int c = i * 256 + tid;
      int row = c >> 5, col = c & 31;            // 32 chunks per 256-col row
      async_g2l_b128(w2base + (unsigned)(row * W2S + col * 8) * 2, W2g + c * 8);
    }
  }

  if (tid < 256) sB1[tid] = b1[tid];
  if (tid < 64) { sB2[tid] = b2[tid]; sAgg[tid] = 0.f; }

  // Build pre_msg rows (overlaps with async weight staging):
  // m in 0..126 real edges, row 127 pad (scale==0 later).
  {
    const float* inB = inputs + (size_t)b * (Nn * NIN);
    int m  = tid >> 1;
    int c0 = (tid & 1) * 32;
    int j  = m + (m >= n ? 1 : 0);
    if (j > 127) j = 127;                        // pad row: any valid node
    const float* srow = inB + j * NIN;
    const float* rrow = inB + n * NIN;
    unsigned short* dst = sPre + m * PS;
#pragma unroll
    for (int c = 0; c < 32; ++c) {
      dst[c0 + c]      = f2bf(srow[c0 + c]);     // senders   -> cols 0..63
      dst[64 + c0 + c] = f2bf(rrow[c0 + c]);     // receivers -> cols 64..127
    }
  }
  wait_async0();                                 // my async copies done
  __syncthreads();                               // everyone's copies visible

  // ---- fc1: 128 -> 256, rows w*16..w*16+15, all 16 channel tiles ----
  v8f acc1[16];
#pragma unroll
  for (int t = 0; t < 16; ++t) acc1[t] = zero8();
  const unsigned short* preRow = sPre + (w * 16 + r) * PS;
#pragma unroll
  for (int kt = 0; kt < 4; ++kt) {
    const unsigned short* bp = preRow + kt * 32 + h * 8;
    v16bf bf = ld2x16(bp, bp + 16);
#pragma unroll
    for (int t = 0; t < 16; ++t) {
      const unsigned short* wrow = sW1 + (t * 16 + r) * W1S + kt * 32 + h * 8;
      acc1[t] = wmma_bf16(ld2x16(wrow, wrow + 16), bf, acc1[t]);
    }
  }
  // bias + relu + pack to bf16x2
  unsigned pk1[16][4];
#pragma unroll
  for (int t = 0; t < 16; ++t)
#pragma unroll
    for (int d = 0; d < 4; ++d) {
      int c = t * 16 + h * 8 + 2 * d;
      float x0 = acc1[t][2 * d]     + sB1[c];
      float x1 = acc1[t][2 * d + 1] + sB1[c + 1];
      pk1[t][d] = pack2(x0 > 0.f ? x0 : 0.f, x1 > 0.f ? x1 : 0.f);
    }

  // ---- fc2: 256 -> 64 (register-resident, no shuffle) ----
  v8f acc2[4];
#pragma unroll
  for (int t = 0; t < 4; ++t) acc2[t] = zero8();
#pragma unroll
  for (int kt = 0; kt < 8; ++kt) {
    v16bf bf = build_bfrag(pk1, kt, h);
#pragma unroll
    for (int t = 0; t < 4; ++t) {
      const unsigned short* wrow = sW2 + (t * 16 + r) * W2S + kt * 32 + h * 8;
      acc2[t] = wmma_bf16(ld2x16(wrow, wrow + 16), bf, acc2[t]);
    }
  }

  // per-row scale = rt0+rt1 (shared edge weights!), then reduce over rows
  int   m     = w * 16 + r;
  float scale = 0.f;
  if (m < 127) {
    size_t e = ((size_t)b * Ee + (size_t)n * 127 + m) * 2;
    scale = rel_types[e] + rel_types[e + 1];
  }
#pragma unroll
  for (int t = 0; t < 4; ++t)
#pragma unroll
    for (int v = 0; v < 8; ++v) {
      int c   = t * 16 + h * 8 + v;
      float x = acc2[t][v] + sB2[c];
      x = (x > 0.f ? x : 0.f) * scale;
#pragma unroll
      for (int mk = 1; mk < 16; mk <<= 1) x += __shfl_xor(x, mk, 32);
      if (r == 0) atomicAdd(&sAgg[c], x);        // ds_add_f32, 32 addrs/half
    }
  __syncthreads();
  if (tid < 64) aggW[((size_t)b * 128 + n) * 64 + tid] = sAgg[tid];
}

// ---------------------------------------------------------------------------
// Kernel 2: node MLP 64 -> 256 -> 256 -> 64 over B*N = 4096 rows.
// grid = 32 blocks of 128 rows, 256 threads. (1.6% of FLOPs; weights stay on
// the global/WGP$ path.)
// ---------------------------------------------------------------------------
__global__ __launch_bounds__(256, 1) void k_node(
    const float* __restrict__ aggW,              // [4096,64]
    const unsigned short* __restrict__ W3g,      // [256,64]
    const unsigned short* __restrict__ W4g,      // [256,256]
    const unsigned short* __restrict__ W5g,      // [64,256]
    const float* __restrict__ b3, const float* __restrict__ b4,
    const float* __restrict__ b5,
    float* __restrict__ out) {                   // [4096,64]
  constexpr int AS = 72;                         // 64 + 8 pad
  __shared__ alignas(16) unsigned short sAct[128 * AS];
  __shared__ float sB1[256];
  __shared__ float sB2[256];
  __shared__ float sB3[64];

  const int blk = blockIdx.x;
  const int tid = threadIdx.x;
  const int lane = tid & 31;
  const int w = tid >> 5, r = lane & 15, h = lane >> 4;

  if (tid < 256) { sB1[tid] = b3[tid]; sB2[tid] = b4[tid]; }
  if (tid < 64) sB3[tid] = b5[tid];
  {
    int lr = tid >> 1, c0 = (tid & 1) * 32;
    const float* src = aggW + ((size_t)blk * 128 + lr) * 64;
    unsigned short* dst = sAct + lr * AS;
#pragma unroll
    for (int c = 0; c < 32; ++c) dst[c0 + c] = f2bf(src[c0 + c]);
  }
  __syncthreads();

  // ---- fc1: 64 -> 256 ----
  v8f acc1[16];
#pragma unroll
  for (int t = 0; t < 16; ++t) acc1[t] = zero8();
  const unsigned short* actRow = sAct + (w * 16 + r) * AS;
#pragma unroll
  for (int kt = 0; kt < 2; ++kt) {
    const unsigned short* bp = actRow + kt * 32 + h * 8;
    v16bf bf = ld2x16(bp, bp + 16);
#pragma unroll
    for (int t = 0; t < 16; ++t) {
      const unsigned short* wrow = W3g + (t * 16 + r) * 64 + kt * 32 + h * 8;
      acc1[t] = wmma_bf16(ld2x16(wrow, wrow + 16), bf, acc1[t]);
    }
  }
  unsigned pk1[16][4];
#pragma unroll
  for (int t = 0; t < 16; ++t)
#pragma unroll
    for (int d = 0; d < 4; ++d) {
      int c = t * 16 + h * 8 + 2 * d;
      float x0 = acc1[t][2 * d]     + sB1[c];
      float x1 = acc1[t][2 * d + 1] + sB1[c + 1];
      pk1[t][d] = pack2(x0 > 0.f ? x0 : 0.f, x1 > 0.f ? x1 : 0.f);
    }

  // ---- fc2: 256 -> 256 ----
  v8f acc2[16];
#pragma unroll
  for (int t = 0; t < 16; ++t) acc2[t] = zero8();
#pragma unroll
  for (int kt = 0; kt < 8; ++kt) {
    v16bf bf = build_bfrag(pk1, kt, h);
#pragma unroll
    for (int t = 0; t < 16; ++t) {
      const unsigned short* wrow = W4g + (t * 16 + r) * 256 + kt * 32 + h * 8;
      acc2[t] = wmma_bf16(ld2x16(wrow, wrow + 16), bf, acc2[t]);
    }
  }
  unsigned pk2[16][4];
#pragma unroll
  for (int t = 0; t < 16; ++t)
#pragma unroll
    for (int d = 0; d < 4; ++d) {
      int c = t * 16 + h * 8 + 2 * d;
      float x0 = acc2[t][2 * d]     + sB2[c];
      float x1 = acc2[t][2 * d + 1] + sB2[c + 1];
      pk2[t][d] = pack2(x0 > 0.f ? x0 : 0.f, x1 > 0.f ? x1 : 0.f);
    }

  // ---- fc3: 256 -> 64 (no relu) ----
  v8f acc3[4];
#pragma unroll
  for (int t = 0; t < 4; ++t) acc3[t] = zero8();
#pragma unroll
  for (int kt = 0; kt < 8; ++kt) {
    v16bf bf = build_bfrag(pk2, kt, h);
#pragma unroll
    for (int t = 0; t < 4; ++t) {
      const unsigned short* wrow = W5g + (t * 16 + r) * 256 + kt * 32 + h * 8;
      acc3[t] = wmma_bf16(ld2x16(wrow, wrow + 16), bf, acc3[t]);
    }
  }
  size_t row = (size_t)blk * 128 + w * 16 + r;
#pragma unroll
  for (int t = 0; t < 4; ++t) {
    float o8[8];
#pragma unroll
    for (int v = 0; v < 8; ++v) o8[v] = acc3[t][v] + sB3[t * 16 + h * 8 + v];
    float4 lo = {o8[0], o8[1], o8[2], o8[3]};
    float4 hi = {o8[4], o8[5], o8[6], o8[7]};
    float* dst = out + row * 64 + t * 16 + h * 8;
    *(float4*)dst       = lo;
    *(float4*)(dst + 4) = hi;
  }
}

// ---------------------------------------------------------------------------
extern "C" void kernel_launch(void* const* d_in, const int* in_sizes, int n_in,
                              void* d_out, int out_size, void* d_ws, size_t ws_size,
                              hipStream_t stream) {
  const float* inputs    = (const float*)d_in[0];
  // d_in[1]=rel_rec, d_in[2]=rel_send: one-hot incidence, replaced by index math
  const float* rel_types = (const float*)d_in[3];
  const float* msg_w1 = (const float*)d_in[4];
  const float* msg_b1 = (const float*)d_in[5];
  const float* msg_w2 = (const float*)d_in[6];
  const float* msg_b2 = (const float*)d_in[7];
  const float* out_w1 = (const float*)d_in[8];
  const float* out_b1 = (const float*)d_in[9];
  const float* out_w2 = (const float*)d_in[10];
  const float* out_b2 = (const float*)d_in[11];
  const float* out_w3 = (const float*)d_in[12];
  const float* out_b3 = (const float*)d_in[13];
  float* pred = (float*)d_out;

  unsigned short* W1g = (unsigned short*)d_ws;   // [256,128] bf16
  unsigned short* W2g = W1g + 32768;             // [64,256]
  unsigned short* W3g = W1g + 49152;             // [256,64]
  unsigned short* W4g = W1g + 65536;             // [256,256]
  unsigned short* W5g = W1g + 131072;            // [64,256]
  float* aggW = (float*)(W1g + 147456);          // [4096,64] f32, 1 MB

  k_convert<<<576, 256, 0, stream>>>(msg_w1, msg_w2, out_w1, out_w2, out_w3, W1g);
  k_edge<<<dim3(Nn, Bsz), 256, 0, stream>>>(inputs, rel_types, W1g, W2g,
                                            msg_b1, msg_b2, aggW);
  k_node<<<32, 256, 0, stream>>>(aggW, W3g, W4g, W5g, out_b1, out_b2, out_b3, pred);
}
